// TRACK_MODEL_28011776705046
// MI455X (gfx1250) — matrix operand order repaired and checked by reading, not compiled, not run on hardware
//
#include <hip/hip_runtime.h>
#include <hip/hip_bf16.h>

typedef __attribute__((ext_vector_type(16))) __bf16 v16bf;
typedef __attribute__((ext_vector_type(8)))  float  v8f;
typedef __bf16 bf16_t;

#define HIDN   256
#define G4     1024
#define BATCH  2048
#define MSTEPS 8
#define HSTEPS 25
#define SALN   512

__device__ __forceinline__ float sigm(float x)   { return 1.0f / (1.0f + __expf(-x)); }
__device__ __forceinline__ float tanh_f(float x) { return 1.0f - 2.0f / (__expf(2.0f * x) + 1.0f); }

// A fragment (16x32 bf16, M x K), loaded from row-major f32 and converted.
// Layout: lanes 0-15 -> M=lane, halves 0..15 = K 0..15;
// lanes 16-31 -> same M, K 16..31 (matches B fragment).
__device__ __forceinline__ v16bf load_a_f32(const float* Atile, long lda, int k) {
  const int lane = threadIdx.x & 31;
  const int m    = lane & 15;
  const int hi   = lane >> 4;
  const float* p0 = Atile + (long)m * lda + k + 16 * hi;
  const float* p1 = p0 + 8;
  const float4 a0 = ((const float4*)p0)[0];
  const float4 a1 = ((const float4*)p0)[1];
  const float4 a2 = ((const float4*)p1)[0];
  const float4 a3 = ((const float4*)p1)[1];
  v16bf r;
  r[0]  = (__bf16)a0.x; r[1]  = (__bf16)a0.y; r[2]  = (__bf16)a0.z; r[3]  = (__bf16)a0.w;
  r[4]  = (__bf16)a1.x; r[5]  = (__bf16)a1.y; r[6]  = (__bf16)a1.z; r[7]  = (__bf16)a1.w;
  r[8]  = (__bf16)a2.x; r[9]  = (__bf16)a2.y; r[10] = (__bf16)a2.z; r[11] = (__bf16)a2.w;
  r[12] = (__bf16)a3.x; r[13] = (__bf16)a3.y; r[14] = (__bf16)a3.z; r[15] = (__bf16)a3.w;
  return r;
}

// B fragment (32x16 bf16, K x N) from weight matrix stored row-major as (N, K)
// (original Wih/Whh layout; B = W.T). Lane n<16: col N=n, K=0..15 contiguous;
// lane n+16: K=16..31. 16 contiguous bf16 = one 32B (2x b128) load.
__device__ __forceinline__ v16bf load_b_bf(const bf16_t* Btile, long ldb, int k) {
  const int lane = threadIdx.x & 31;
  const int n    = lane & 15;
  const int hi   = lane >> 4;
  const bf16_t* p = Btile + (long)n * ldb + k + 16 * hi;
  return *(const v16bf*)p;
}

// One GEMM segment over K, accumulating into NT live n-tile accumulators.
// k is the OUTER loop: one A fragment per k-step feeds NT independent WMMAs,
// breaking the WMMA->WMMA RAW chain and amortizing the f32->bf16 A conversion.
template <int NT>
__device__ __forceinline__ void gemm_seg(v8f* acc, const float* Atile, long lda,
                                         const bf16_t* Wbase, long ldw, int K,
                                         int nbase) {
  for (int k = 0; k < K; k += 32) {
    const v16bf a = load_a_f32(Atile, lda, k);
#pragma unroll
    for (int tt = 0; tt < NT; ++tt) {
      const v16bf b = load_b_bf(Wbase + (long)(nbase + tt * 16) * ldw, ldw, k);
      acc[tt] = __builtin_amdgcn_wmma_f32_16x16x32_bf16(false, a, false, b, (short)0,
                                                        acc[tt], false, false);
    }
  }
}

// Fused LSTM cell: gates = bias + [posX@posW.T] + X1@W1.T + X2@W2.T + H@Whh.T,
// then activations; updates C and H in place (row m depends only on H row m).
// One workgroup = 16 batch rows; 8 waves x 8 live n-tiles cover all 1024 gate
// columns. Gates staged in LDS for the activation pass.
__global__ __launch_bounds__(256) void lstm_cell_kernel(
    const float* X1, long ldx1, const bf16_t* W1, long ldw1, int K1,
    const float* X2, long ldx2, const bf16_t* W2, long ldw2, int K2,
    const float* H, const bf16_t* Whh,
    const float* posX, long ldpx, const float* posW,
    const float* bias, float* Cst, float* Hst)
{
  __shared__ float gates[16][G4];
  const int m0    = blockIdx.x * 16;
  const int wave  = threadIdx.x >> 5;
  const int lane  = threadIdx.x & 31;
  const int nlo   = lane & 15;
  const int hi    = lane >> 4;
  const int nbase = wave * 128;

  v8f acc[8];
#pragma unroll
  for (int tt = 0; tt < 8; ++tt) {
    const float bv = bias[nbase + tt * 16 + nlo];
#pragma unroll
    for (int r = 0; r < 8; ++r) acc[tt][r] = bv;
  }

  if (posW) {  // tiny K=3 input: scalar MACs folded into the WMMA accumulator layout
    float x0[8], x1[8], x2[8];
#pragma unroll
    for (int r = 0; r < 8; ++r) {
      const long m = m0 + r + 8 * hi;
      x0[r] = posX[m * ldpx + 0];
      x1[r] = posX[m * ldpx + 1];
      x2[r] = posX[m * ldpx + 2];
    }
#pragma unroll
    for (int tt = 0; tt < 8; ++tt) {
      const int n = nbase + tt * 16 + nlo;
      const float w0 = posW[n * 3 + 0];
      const float w1 = posW[n * 3 + 1];
      const float w2 = posW[n * 3 + 2];
#pragma unroll
      for (int r = 0; r < 8; ++r) acc[tt][r] += x0[r] * w0 + x1[r] * w1 + x2[r] * w2;
    }
  }

  if (K1 > 0) gemm_seg<8>(acc, X1 + (long)m0 * ldx1, ldx1, W1, ldw1, K1, nbase);
  if (K2 > 0) gemm_seg<8>(acc, X2 + (long)m0 * ldx2, ldx2, W2, ldw2, K2, nbase);
  gemm_seg<8>(acc, H + (long)m0 * HIDN, HIDN, Whh, HIDN, HIDN, nbase);

#pragma unroll
  for (int tt = 0; tt < 8; ++tt)
#pragma unroll
    for (int r = 0; r < 8; ++r) gates[r + 8 * hi][nbase + tt * 16 + nlo] = acc[tt][r];
  __syncthreads();

  for (int idx = threadIdx.x; idx < 16 * HIDN; idx += 256) {
    const int row = idx >> 8;
    const int col = idx & 255;
    const float gi = gates[row][col];
    const float gf = gates[row][col + 256];
    const float gg = gates[row][col + 512];
    const float go = gates[row][col + 768];
    const long gm  = m0 + row;
    float c = Cst[gm * HIDN + col];
    c = sigm(gf) * c + sigm(gi) * tanh_f(gg);
    const float h = sigm(go) * tanh_f(c);
    Cst[gm * HIDN + col] = c;
    Hst[gm * HIDN + col] = h;
  }
}

// Decoder head: fused = hf@fc1W.T + fc1b (WMMA, 2 live tiles/wave),
// delta = fused@fcoW.T + fcob (scalar, N=3), pred = dec_in + delta;
// writes pred to output[:, t, :] and back into dec_in (autoregressive feed).
__global__ __launch_bounds__(256) void fc_head_kernel(
    const float* Hf, const bf16_t* W1bf, const float* b1,
    const float* fcoW, const float* fcob,
    float* dec_in, float* out, int t)
{
  __shared__ float fused[16][HIDN];
  const int m0    = blockIdx.x * 16;
  const int wave  = threadIdx.x >> 5;
  const int lane  = threadIdx.x & 31;
  const int nlo   = lane & 15;
  const int hi    = lane >> 4;
  const int nbase = wave * 32;

  v8f acc[2];
#pragma unroll
  for (int tt = 0; tt < 2; ++tt) {
    const float bv = b1[nbase + tt * 16 + nlo];
#pragma unroll
    for (int r = 0; r < 8; ++r) acc[tt][r] = bv;
  }
  gemm_seg<2>(acc, Hf + (long)m0 * HIDN, HIDN, W1bf, HIDN, HIDN, nbase);
#pragma unroll
  for (int tt = 0; tt < 2; ++tt)
#pragma unroll
    for (int r = 0; r < 8; ++r) fused[r + 8 * hi][nbase + tt * 16 + nlo] = acc[tt][r];
  __syncthreads();

  if (threadIdx.x < 48) {
    const int row = threadIdx.x / 3;
    const int j   = threadIdx.x % 3;
    float s = fcob[j];
    for (int k = 0; k < HIDN; ++k) s += fused[row][k] * fcoW[j * HIDN + k];
    const long gm = m0 + row;
    const float pred = dec_in[gm * 3 + j] + s;
    dec_in[gm * 3 + j] = pred;
    out[(gm * HSTEPS + t) * 3 + j] = pred;
  }
}

__global__ void f2bf_kernel(bf16_t* dst, const float* src, int n) {
  const int i = blockIdx.x * 256 + threadIdx.x;
  if (i < n) dst[i] = (__bf16)src[i];
}
__global__ void addvec_kernel(float* dst, const float* a, const float* b, int n) {
  const int i = blockIdx.x * 256 + threadIdx.x;
  if (i < n) dst[i] = a[i] + b[i];
}
__global__ void zero_kernel(float* p, long n) {
  const long i = (long)blockIdx.x * 256 + threadIdx.x;
  if (i < n) p[i] = 0.0f;
}
__global__ void copy_kernel(float* dst, const float* src, int n) {
  const int i = blockIdx.x * 256 + threadIdx.x;
  if (i < n) dst[i] = src[i];
}

extern "C" void kernel_launch(void* const* d_in, const int* in_sizes, int n_in,
                              void* d_out, int out_size, void* d_ws, size_t ws_size,
                              hipStream_t stream)
{
  (void)in_sizes; (void)out_size;
  if (n_in < 32) return;

  const float* enc_pos = (const float*)d_in[0];   // (B, 8, 3)
  const float* enc_sal = (const float*)d_in[1];   // (B, 8, 512)
  const float* dec_pos = (const float*)d_in[2];   // (B, 1, 3)
  const float* dec_sal = (const float*)d_in[3];   // (B, 25, 512)
  struct LP { const float *Wih, *Whh, *bih, *bhh; };
  LP lp[6];  // 0 pos_enc, 1 sal_enc, 2 fuse_enc, 3 pos_dec, 4 sal_dec, 5 fuse_dec
  for (int i = 0; i < 6; ++i) {
    lp[i].Wih = (const float*)d_in[4 + i * 4 + 0];
    lp[i].Whh = (const float*)d_in[4 + i * 4 + 1];
    lp[i].bih = (const float*)d_in[4 + i * 4 + 2];
    lp[i].bhh = (const float*)d_in[4 + i * 4 + 3];
  }
  const float* fc1W = (const float*)d_in[28];
  const float* fc1b = (const float*)d_in[29];
  const float* fcoW = (const float*)d_in[30];
  const float* fcob = (const float*)d_in[31];
  float* out = (float*)d_out;

  // ---- workspace carve-out (256B aligned) ----
  char* ws = (char*)d_ws;
  size_t off = 0;
  auto alloc = [&](size_t bytes) -> char* {
    char* p = ws + off;
    off += (bytes + 255) & ~(size_t)255;
    return p;
  };
  bf16_t* whh_bf[6];
  for (int i = 0; i < 6; ++i) whh_bf[i] = (bf16_t*)alloc((size_t)G4 * HIDN * 2);
  bf16_t* wih_bf[6] = {nullptr, nullptr, nullptr, nullptr, nullptr, nullptr};
  for (int i : {1, 2, 4, 5}) wih_bf[i] = (bf16_t*)alloc((size_t)G4 * SALN * 2);
  bf16_t* fc1bf = (bf16_t*)alloc((size_t)HIDN * HIDN * 2);
  float* biasc[6];
  for (int i = 0; i < 6; ++i) biasc[i] = (float*)alloc(G4 * 4);
  float* states = (float*)alloc((size_t)6 * BATCH * HIDN * 4);  // H0..H2, C0..C2
  float* Hs[3], *Cs[3];
  for (int i = 0; i < 3; ++i) {
    Hs[i] = states + (size_t)i * BATCH * HIDN;
    Cs[i] = states + (size_t)(3 + i) * BATCH * HIDN;
  }
  float* dec_in = (float*)alloc((size_t)BATCH * 3 * 4);
  if (off > ws_size) return;

  // ---- per-launch prep: weight conversion, bias combine, state init ----
  for (int i = 0; i < 6; ++i) {
    const int n = G4 * HIDN;
    f2bf_kernel<<<(n + 255) / 256, 256, 0, stream>>>(whh_bf[i], lp[i].Whh, n);
  }
  for (int i : {1, 2, 4, 5}) {
    const int n = G4 * SALN;
    f2bf_kernel<<<(n + 255) / 256, 256, 0, stream>>>(wih_bf[i], lp[i].Wih, n);
  }
  {
    const int n = HIDN * HIDN;
    f2bf_kernel<<<(n + 255) / 256, 256, 0, stream>>>(fc1bf, fc1W, n);
  }
  for (int i = 0; i < 6; ++i)
    addvec_kernel<<<(G4 + 255) / 256, 256, 0, stream>>>(biasc[i], lp[i].bih, lp[i].bhh, G4);
  {
    const long n = (long)6 * BATCH * HIDN;
    zero_kernel<<<(unsigned)((n + 255) / 256), 256, 0, stream>>>(states, n);
  }
  copy_kernel<<<(BATCH * 3 + 255) / 256, 256, 0, stream>>>(dec_in, dec_pos, BATCH * 3);

  const dim3 grid(BATCH / 16), blk(256);

  // ---- encoder: 8 steps x (pos, sal, fuse) ----
  for (int t = 0; t < MSTEPS; ++t) {
    lstm_cell_kernel<<<grid, blk, 0, stream>>>(
        nullptr, 0, nullptr, 0, 0, nullptr, 0, nullptr, 0, 0,
        Hs[0], whh_bf[0],
        enc_pos + (long)t * 3, (long)(MSTEPS * 3), lp[0].Wih,
        biasc[0], Cs[0], Hs[0]);
    lstm_cell_kernel<<<grid, blk, 0, stream>>>(
        enc_sal + (long)t * SALN, (long)(MSTEPS * SALN), wih_bf[1], (long)SALN, SALN,
        nullptr, 0, nullptr, 0, 0,
        Hs[1], whh_bf[1], nullptr, 0, nullptr,
        biasc[1], Cs[1], Hs[1]);
    lstm_cell_kernel<<<grid, blk, 0, stream>>>(             // conc = [h_sal, h_pos]
        Hs[1], (long)HIDN, wih_bf[2], (long)(2 * HIDN), HIDN,
        Hs[0], (long)HIDN, wih_bf[2] + HIDN, (long)(2 * HIDN), HIDN,
        Hs[2], whh_bf[2], nullptr, 0, nullptr,
        biasc[2], Cs[2], Hs[2]);
  }

  // ---- decoder: 25 steps x (pos, sal, fuse, head); states continue from encoder ----
  for (int t = 0; t < HSTEPS; ++t) {
    lstm_cell_kernel<<<grid, blk, 0, stream>>>(
        nullptr, 0, nullptr, 0, 0, nullptr, 0, nullptr, 0, 0,
        Hs[0], whh_bf[3],
        dec_in, (long)3, lp[3].Wih,
        biasc[3], Cs[0], Hs[0]);
    lstm_cell_kernel<<<grid, blk, 0, stream>>>(
        dec_sal + (long)t * SALN, (long)(HSTEPS * SALN), wih_bf[4], (long)SALN, SALN,
        nullptr, 0, nullptr, 0, 0,
        Hs[1], whh_bf[4], nullptr, 0, nullptr,
        biasc[4], Cs[1], Hs[1]);
    lstm_cell_kernel<<<grid, blk, 0, stream>>>(
        Hs[1], (long)HIDN, wih_bf[5], (long)(2 * HIDN), HIDN,
        Hs[0], (long)HIDN, wih_bf[5] + HIDN, (long)(2 * HIDN), HIDN,
        Hs[2], whh_bf[5], nullptr, 0, nullptr,
        biasc[5], Cs[2], Hs[2]);
    fc_head_kernel<<<grid, blk, 0, stream>>>(Hs[2], fc1bf, fc1b, fcoW, fcob, dec_in, out, t);
  }
}
